// ScaleDotProductAttention_56513179680978
// MI455X (gfx1250) — matrix operand order repaired and checked by reading, not compiled, not run on hardware
//
#include <hip/hip_runtime.h>

#define BB 2
#define HH 16
#define LL 2048
#define DD 64

typedef __attribute__((ext_vector_type(16))) __bf16 v16bf;
typedef __attribute__((ext_vector_type(8)))  float  v8f;
typedef __attribute__((ext_vector_type(4)))  unsigned int u32x4;
typedef __attribute__((ext_vector_type(8)))  int    i32x8;
typedef __attribute__((ext_vector_type(4)))  int    i32x4;

__device__ __forceinline__ v8f wmma_bf16(v16bf a, v16bf b, v8f c) {
  // D = A(16x32 bf16) * B(32x16 bf16) + C(16x16 f32)
  return __builtin_amdgcn_wmma_f32_16x16x32_bf16(false, a, false, b, (short)0, c,
                                                 false, false);
}

// One workgroup = 16 wave32 waves; wave h owns head h.
// Workgroup owns (b, i0..i0+15). Iterates over j in chunks of 32.
__global__ __launch_bounds__(512, 1)
void attn_headsoftmax_fused(const float* __restrict__ q, const float* __restrict__ k,
                            const float* __restrict__ v, const int* __restrict__ mask,
                            float* __restrict__ out, float* __restrict__ score) {
  __shared__ float lds_p[HH][16][32];  // 32 KB: per-head 16x32 logits -> probs
  __shared__ int   lds_mask[16][32];   // 2 KB: mask tile (TDM destination)

  const int tid  = threadIdx.x;
  const int lane = tid & 31;
  const int wid  = tid >> 5;           // head index for this wave
  const int blk  = blockIdx.x;         // b*128 + itile
  const int bb   = blk >> 7;
  const int i0   = (blk & 127) << 4;

  const int m   = lane & 15;           // M row (A layout)
  const int hi  = lane >> 4;           // lane half
  const int kb  = hi * 16;             // A-layout K base (16-bit A 16x32, contiguous k = 16h + s)
  const int n   = lane & 15;           // N col (B/C layouts)
  const int kk0 = hi * 16;             // B-layout K base

  // ---- A matrices: Q tile (rows i0..i0+15, all 64 K) in WMMA A layout, bf16
  const float* qrow = q + (((size_t)bb * HH + wid) * LL + (size_t)(i0 + m)) * DD;
  v16bf aq0, aq1;
  #pragma unroll
  for (int e = 0; e < 8; ++e) {
    aq0[e]     = (__bf16)qrow[kb + e];
    aq0[e + 8] = (__bf16)qrow[kb + 8 + e];
    aq1[e]     = (__bf16)qrow[kb + 32 + e];
    aq1[e + 8] = (__bf16)qrow[kb + 40 + e];
  }

  // K viewed as row-major [64][2048] (the reference's reshape quirk), V as [2048][64]
  const float* kbase = k + ((size_t)bb * HH + wid) * (size_t)LL * DD;
  const float* vbase = v + ((size_t)bb * HH + wid) * (size_t)LL * DD;
  float* outbase     = out + ((size_t)bb * HH + wid) * (size_t)LL * DD;

  v8f acc0 = {}, acc1 = {}, acc2 = {}, acc3 = {};

  const unsigned lds_mask_off = (unsigned)(unsigned long long)(void*)&lds_mask[0][0];

  for (int j0 = 0; j0 < LL; j0 += 32) {
    // ---- Tensor Data Mover: fetch 16x32 int32 mask tile into LDS (wave 0 only)
    if (wid == 0) {
#if defined(__has_builtin) && __has_builtin(__builtin_amdgcn_tensor_load_to_lds)
      const unsigned long long ga = (unsigned long long)(const void*)
          (mask + ((size_t)bb * LL + i0) * LL + j0);
      u32x4 g0; i32x8 g1; i32x4 g2 = {0,0,0,0}; i32x4 g3 = {0,0,0,0};
      i32x8 g4 = {0,0,0,0,0,0,0,0};
      g0[0] = 1u;                                               // count=1, user mode
      g0[1] = lds_mask_off;                                     // lds_addr (bytes)
      g0[2] = (unsigned)(ga & 0xFFFFFFFFu);                     // global_addr[31:0]
      g0[3] = (unsigned)((ga >> 32) & 0x1FFFFFFu) | (2u << 30); // addr[56:32] | type=2
      g1[0] = (int)0x00020000;                                  // data_size=4B code 2
      g1[1] = (int)((LL & 0xFFFF) << 16);                       // tensor_dim0 lo16
      g1[2] = (int)(((LL >> 16) & 0xFFFF) | ((LL & 0xFFFF) << 16)); // dim0 hi | dim1 lo
      g1[3] = (int)(((LL >> 16) & 0xFFFF) | (32u << 16));       // dim1 hi | tile_dim0=32
      g1[4] = 16;                                               // tile_dim1=16, tile_dim2=0
      g1[5] = LL;                                               // tensor_dim0_stride lo32
      g1[6] = 0;                                                // stride0 hi | stride1 lo16
      g1[7] = (int)(((size_t)LL * (size_t)LL) >> 16);           // tensor_dim1_stride hi
      __builtin_amdgcn_tensor_load_to_lds(g0, g1, g2, g3, g4, 0);
#else
      if (lane < 16) {
        #pragma unroll 4
        for (int jj = 0; jj < 32; ++jj)
          lds_mask[lane][jj] = mask[((size_t)bb * LL + i0 + lane) * LL + j0 + jj];
      }
#endif
    }

    // Prefetch next chunk of K / V while we compute (CDNA5 global_prefetch_b8)
    if (j0 + 32 < LL) {
      __builtin_prefetch(kbase + (size_t)kk0 * LL + (j0 + 32) + n, 0, 1);
      __builtin_prefetch(vbase + (size_t)(j0 + 32 + kk0) * DD + n, 0, 1);
    }

    // ---- QK^T logits: two 16-wide column subtiles (bf16 WMMA, K=64 in 2 steps)
    v8f s0 = {}, s1 = {};
    {
      const float* kc = kbase + (size_t)kk0 * LL + (j0 + n);
      v16bf bk0, bk1;
      #pragma unroll
      for (int e = 0; e < 16; ++e) bk0[e] = (__bf16)kc[(size_t)e * LL];
      #pragma unroll
      for (int e = 0; e < 16; ++e) bk1[e] = (__bf16)kc[(size_t)(e + 32) * LL];
      s0 = wmma_bf16(aq0, bk0, s0);
      s0 = wmma_bf16(aq1, bk1, s0);
      #pragma unroll
      for (int e = 0; e < 16; ++e) bk0[e] = (__bf16)kc[(size_t)e * LL + 16];
      #pragma unroll
      for (int e = 0; e < 16; ++e) bk1[e] = (__bf16)kc[(size_t)(e + 32) * LL + 16];
      s1 = wmma_bf16(aq0, bk0, s1);
      s1 = wmma_bf16(aq1, bk1, s1);
    }

    // Scale by 1/sqrt(64) and publish logits to LDS (C layout -> row major)
    #pragma unroll
    for (int r = 0; r < 8; ++r) {
      lds_p[wid][hi * 8 + r][n]      = s0[r] * 0.125f;
      lds_p[wid][hi * 8 + r][n + 16] = s1[r] * 0.125f;
    }
#if defined(__has_builtin) && __has_builtin(__builtin_amdgcn_s_wait_tensorcnt)
    if (wid == 0) __builtin_amdgcn_s_wait_tensorcnt(0);
#endif
    __syncthreads();

    // ---- Softmax across the 16 HEADS (reference quirk 2), one thread per (i,j)
    {
      const int si = wid;      // i row 0..15
      const int sj = lane;     // j col 0..31
      float* sc = score + ((size_t)bb * HH * LL + (size_t)(i0 + si)) * LL
                        + (size_t)(j0 + sj);
      const int mk = lds_mask[si][sj];
      if (mk == 0) {
        // all heads masked to -1e12 -> uniform 1/16
        #pragma unroll
        for (int hh = 0; hh < HH; ++hh) {
          lds_p[hh][si][sj] = 0.0625f;
          sc[(size_t)hh * LL * LL] = 0.0625f;
        }
      } else {
        float mx = lds_p[0][si][sj];
        #pragma unroll
        for (int hh = 1; hh < HH; ++hh) mx = fmaxf(mx, lds_p[hh][si][sj]);
        float ex[HH];
        float sum = 0.f;
        #pragma unroll
        for (int hh = 0; hh < HH; ++hh) {
          ex[hh] = __expf(lds_p[hh][si][sj] - mx);
          sum += ex[hh];
        }
        const float inv = __builtin_amdgcn_rcpf(sum);
        #pragma unroll
        for (int hh = 0; hh < HH; ++hh) {
          const float p = ex[hh] * inv;
          lds_p[hh][si][sj] = p;
          sc[(size_t)hh * LL * LL] = p;
        }
      }
    }
    __syncthreads();

    // ---- P @ V: A = prob tile (16x32) from LDS, B = V rows j0..j0+31, 4 d-tiles
    v16bf ap;
    #pragma unroll
    for (int e = 0; e < 8; ++e) {
      ap[e]     = (__bf16)lds_p[wid][m][kb + e];
      ap[e + 8] = (__bf16)lds_p[wid][m][kb + 8 + e];
    }
    {
      const float* vc = vbase + (size_t)(j0 + kk0) * DD + n;
      v16bf bv;
      #pragma unroll
      for (int e = 0; e < 16; ++e) bv[e] = (__bf16)vc[(size_t)e * DD];
      acc0 = wmma_bf16(ap, bv, acc0);
      #pragma unroll
      for (int e = 0; e < 16; ++e) bv[e] = (__bf16)vc[(size_t)e * DD + 16];
      acc1 = wmma_bf16(ap, bv, acc1);
      #pragma unroll
      for (int e = 0; e < 16; ++e) bv[e] = (__bf16)vc[(size_t)e * DD + 32];
      acc2 = wmma_bf16(ap, bv, acc2);
      #pragma unroll
      for (int e = 0; e < 16; ++e) bv[e] = (__bf16)vc[(size_t)e * DD + 48];
      acc3 = wmma_bf16(ap, bv, acc3);
    }
    __syncthreads();  // LDS (logits + mask tile) reused next iteration
  }

  // ---- Write out tile: C layout, N = lane&15, row M = r + 8*hi
  #pragma unroll
  for (int r = 0; r < 8; ++r) {
    const size_t row = (size_t)(i0 + hi * 8 + r) * DD;
    outbase[row + n]      = acc0[r];
    outbase[row + 16 + n] = acc1[r];
    outbase[row + 32 + n] = acc2[r];
    outbase[row + 48 + n] = acc3[r];
  }
}

extern "C" void kernel_launch(void* const* d_in, const int* in_sizes, int n_in,
                              void* d_out, int out_size, void* d_ws, size_t ws_size,
                              hipStream_t stream) {
  (void)in_sizes; (void)n_in; (void)out_size; (void)d_ws; (void)ws_size;
  const float* q    = (const float*)d_in[0];
  const float* k    = (const float*)d_in[1];
  const float* v    = (const float*)d_in[2];
  const int*   mask = (const int*)d_in[3];
  float* out   = (float*)d_out;
  float* score = out + (size_t)BB * HH * LL * DD;  // outputs concatenated: out, score

  dim3 grid(BB * (LL / 16));   // 256 workgroups
  dim3 block(512);             // 16 wave32 waves = 16 heads
  hipLaunchKernelGGL(attn_headsoftmax_fused, grid, block, 0, stream,
                     q, k, v, mask, out, score);
}
